// CoherentMamba_47811575939686
// MI455X (gfx1250) — matrix operand order repaired and checked by reading, not compiled, not run on hardware
//
#include <hip/hip_runtime.h>
#include <math.h>

#define D_MODEL   512
#define N_LAYERS  4
#define D_STATE   16
#define D_CONV    4
#define D_INNER   1024
#define DT_RANK   32
#define N_CLASSES 4
#define IN_CH     2
#define BATCH     2
#define SEQLEN    2048
#define M_TOT     (BATCH*SEQLEN)   /* 4096 token rows */

typedef __bf16        v16bf __attribute__((ext_vector_type(16)));
typedef float         v8f   __attribute__((ext_vector_type(8)));
typedef unsigned int  v8u   __attribute__((ext_vector_type(8)));

__device__ __forceinline__ unsigned short f32_to_bf16(float f) {
  unsigned int u = __builtin_bit_cast(unsigned int, f);
  unsigned int r = u + 0x7FFFu + ((u >> 16) & 1u);   // round-to-nearest-even
  return (unsigned short)(r >> 16);
}
__device__ __forceinline__ unsigned int pack2_bf16(float lo, float hi) {
  return ((unsigned int)f32_to_bf16(hi) << 16) | (unsigned int)f32_to_bf16(lo);
}
__device__ __forceinline__ float siluf(float x)     { return x / (1.f + __expf(-x)); }
__device__ __forceinline__ float softplusf(float x) { return fmaxf(x, 0.f) + log1pf(__expf(-fabsf(x))); }

// ---------------------------------------------------------------------------
// WMMA GEMM:  C[M,N] = A[M,K] * W[N,K]^T   (+ epilogue)
//   mode 0: none   mode 1: softplus(acc + bias[n])   mode 2: acc + res[m,n]
// REQUIRES: M % 256 == 0, N % 64 == 0, K % 32 == 0, 16B-aligned rows
// (true for every projection in this model) -> zero bounds checks.
// Block: 256 threads = 8 waves. Tile: BM=256, BN=64, BK=32.
// Wave w owns a 32x64 strip: 2 M-subtiles x 4 N-subtiles = 8 WMMA/K-step,
// reusing each B fragment across 2 WMMAs and each A fragment across 4.
// ---------------------------------------------------------------------------
#define BM 256
#define BN 64
#define BK 32
#define LSTR32 17   // dwords per LDS row (odd stride -> conflict-friendly)

__global__ __launch_bounds__(256)
void gemm_bf16_wmma(const float* __restrict__ A, int lda,
                    const float* __restrict__ W, int ldw,
                    float* __restrict__ C, int ldc,
                    int K, int mode,
                    const float* __restrict__ bias,
                    const float* __restrict__ res)
{
  __shared__ unsigned int XsD[BM * LSTR32];   // 17408 B
  __shared__ unsigned int WsD[BN * LSTR32];   //  4352 B

  const int tid  = threadIdx.x;
  const int wave = tid >> 5;
  const int lane = tid & 31;
  const int hf   = lane >> 4;   // half-wave select
  const int l16  = lane & 15;
  const int m0   = blockIdx.y * BM;
  const int n0   = blockIdx.x * BN;

  v8f acc[2][4] = {};

  for (int k0 = 0; k0 < K; k0 += BK) {
    // ---- stage A tile (256x32) as packed bf16 pairs: 8 float4 per thread ----
#pragma unroll
    for (int it = 0; it < 8; ++it) {
      int g = tid + it * 256;            // 2048 float4 groups (8 per row)
      int r = g >> 3, c = (g & 7) * 4;
      const float4 f = *(const float4*)(A + (long)(m0 + r) * lda + k0 + c);
      XsD[r * LSTR32 + (c >> 1)    ] = pack2_bf16(f.x, f.y);
      XsD[r * LSTR32 + (c >> 1) + 1] = pack2_bf16(f.z, f.w);
    }
    // ---- stage W tile (64x32): 2 float4 per thread ----
#pragma unroll
    for (int it = 0; it < 2; ++it) {
      int g = tid + it * 256;            // 512 float4 groups
      int r = g >> 3, c = (g & 7) * 4;
      const float4 f = *(const float4*)(W + (long)(n0 + r) * ldw + k0 + c);
      WsD[r * LSTR32 + (c >> 1)    ] = pack2_bf16(f.x, f.y);
      WsD[r * LSTR32 + (c >> 1) + 1] = pack2_bf16(f.z, f.w);
    }
    // best-effort prefetch of the next K tile while this one computes
    if (k0 + BK < K) {
      int g = tid;
      int r = g >> 3, c = (g & 7) * 4;
      __builtin_prefetch(A + (long)(m0 + r) * lda + k0 + BK + c, 0, 1);
      __builtin_prefetch(A + (long)(m0 + r + 128) * lda + k0 + BK + c, 0, 1);
      if (g < 512) __builtin_prefetch(W + (long)(n0 + r) * ldw + k0 + BK + c, 0, 1);
    }
    __syncthreads();

    // ---- A fragments (2 M-subtiles): ISA interleaved-K layout ----
    v16bf aF[2];
#pragma unroll
    for (int mt = 0; mt < 2; ++mt) {
      const int mrow = wave * 32 + mt * 16 + l16;
      v8u aU;
#pragma unroll
      for (int j = 0; j < 8; ++j) {
        // dword j covers K pair {kk, kk+1}
        int kk = (j < 4) ? (2 * j + hf * 8) : (2 * j + 8 + hf * 8);
        aU[j] = XsD[mrow * LSTR32 + (kk >> 1)];
      }
      aF[mt] = __builtin_bit_cast(v16bf, aU);
    }

    // ---- 4 N-subtiles x 2 M-subtiles = 8 WMMA; B frag same interleaved-K as A ----
#pragma unroll
    for (int nt = 0; nt < 4; ++nt) {
      const int nrow = nt * 16 + l16;
      v8u bU;
#pragma unroll
      for (int j = 0; j < 8; ++j)
        bU[j] = WsD[nrow * LSTR32 + ((j < 4) ? (j + hf * 4) : (j + 4 + hf * 4))];   // K pair matches A: kk = 2j + 8*hf (+8 if j>=4)
      v16bf bF = __builtin_bit_cast(v16bf, bU);
#pragma unroll
      for (int mt = 0; mt < 2; ++mt)
        acc[mt][nt] = __builtin_amdgcn_wmma_f32_16x16x32_bf16(
            false, aF[mt], false, bF, (short)0, acc[mt][nt], false, false);
    }
    __syncthreads();
  }

  // ---- epilogue: D layout  M = r + 8*hf within each 16-row subtile ----
#pragma unroll
  for (int mt = 0; mt < 2; ++mt) {
#pragma unroll
    for (int nt = 0; nt < 4; ++nt) {
      int n = n0 + nt * 16 + l16;
#pragma unroll
      for (int r = 0; r < 8; ++r) {
        int m = m0 + wave * 32 + mt * 16 + r + hf * 8;
        float v = acc[mt][nt][r];
        if (mode == 1)      v = softplusf(v + bias[n]);
        else if (mode == 2) v += res[(long)m * ldc + n];
        C[(long)m * ldc + n] = v;
      }
    }
  }
}

// ---------------------------------------------------------------------------
// h[m,d] = x[b,0,l]*w[d,0] + x[b,1,l]*w[d,1] + b[d]
// ---------------------------------------------------------------------------
__global__ __launch_bounds__(256)
void embed_kernel(const float* __restrict__ x, const float* __restrict__ w,
                  const float* __restrict__ bb, float* __restrict__ h)
{
  int idx = blockIdx.x * 256 + threadIdx.x;   // m*512 + d
  if (idx >= M_TOT * D_MODEL) return;
  int d = idx & (D_MODEL - 1);
  int m = idx >> 9;
  int b = m >> 11, l = m & (SEQLEN - 1);
  const float* xb = x + (long)b * IN_CH * SEQLEN;
  h[idx] = xb[l] * w[d * 2] + xb[SEQLEN + l] * w[d * 2 + 1] + bb[d];
}

// ---------------------------------------------------------------------------
// LayerNorm over D_MODEL=512, one block per row, 256 threads, 2 elems/thread
// ---------------------------------------------------------------------------
__global__ __launch_bounds__(256)
void layernorm_kernel(const float* __restrict__ x, const float* __restrict__ w,
                      const float* __restrict__ bb, float* __restrict__ o)
{
  __shared__ float s1[256], s2[256];
  int m = blockIdx.x, tid = threadIdx.x;
  const float* row = x + (long)m * D_MODEL;
  float v0 = row[tid], v1 = row[tid + 256];
  s1[tid] = v0 + v1;
  s2[tid] = v0 * v0 + v1 * v1;
  __syncthreads();
  for (int s = 128; s > 0; s >>= 1) {
    if (tid < s) { s1[tid] += s1[tid + s]; s2[tid] += s2[tid + s]; }
    __syncthreads();
  }
  float mean = s1[0] * (1.f / D_MODEL);
  float var  = s2[0] * (1.f / D_MODEL) - mean * mean;
  float rs   = rsqrtf(var + 1e-5f);
  float* orow = o + (long)m * D_MODEL;
  orow[tid]       = (v0 - mean) * rs * w[tid]       + bb[tid];
  orow[tid + 256] = (v1 - mean) * rs * w[tid + 256] + bb[tid + 256];
}

// ---------------------------------------------------------------------------
// Depthwise causal conv (width 4) + SiLU.  xx = xz[:, 0:1024] (row stride 2048)
// ---------------------------------------------------------------------------
__global__ __launch_bounds__(256)
void conv_silu_kernel(const float* __restrict__ xz, const float* __restrict__ cw,
                      const float* __restrict__ cb, float* __restrict__ xc)
{
  int idx = blockIdx.x * 256 + threadIdx.x;   // m*1024 + d
  if (idx >= M_TOT * D_INNER) return;
  int d = idx & (D_INNER - 1);
  int m = idx >> 10;
  int l = m & (SEQLEN - 1);
  float a = cb[d];
#pragma unroll
  for (int k = 0; k < D_CONV; ++k) {
    int li = l + k - (D_CONV - 1);
    if (li >= 0)
      a += xz[(long)(m + k - (D_CONV - 1)) * (2 * D_INNER) + d] * cw[d * D_CONV + k];
  }
  xc[idx] = siluf(a);
}

// ---------------------------------------------------------------------------
// Selective scan: one thread per (b,d) channel, 16-state in registers.
// B/C staged in LDS for 32 timesteps per barrier (all threads of a block
// share the same batch index b). Fuses y = (scan + D*u) * silu(z).
// ---------------------------------------------------------------------------
#define SCHUNK 32

__global__ __launch_bounds__(256)
void scan_kernel(const float* __restrict__ dt, const float* __restrict__ x_dbl,
                 const float* __restrict__ xc, const float* __restrict__ xz,
                 const float* __restrict__ A_log, const float* __restrict__ Dp,
                 float* __restrict__ y)
{
  __shared__ float BCs[SCHUNK * 2 * D_STATE];   // 32 steps x 32 floats
  int t = blockIdx.x * 256 + threadIdx.x;       // 0..2047
  int b = t >> 10, d = t & (D_INNER - 1);
  float a[D_STATE], hst[D_STATE];
#pragma unroll
  for (int n = 0; n < D_STATE; ++n) {
    a[n] = -__expf(A_log[d * D_STATE + n]);
    hst[n] = 0.f;
  }
  float Dv = Dp[d];

  for (int c0 = 0; c0 < SEQLEN; c0 += SCHUNK) {
    // stage B/C for SCHUNK timesteps: 1024 floats, 4 per thread
#pragma unroll
    for (int j = 0; j < (SCHUNK * 2 * D_STATE) / 256; ++j) {
      int e = threadIdx.x + j * 256;
      int st = e >> 5, q = e & 31;
      BCs[e] = x_dbl[(long)(b * SEQLEN + c0 + st) * 64 + DT_RANK + q];
    }
    __syncthreads();

    for (int s = 0; s < SCHUNK; ++s) {
      int m = b * SEQLEN + c0 + s;
      const float* bc = BCs + s * 2 * D_STATE;
      float dtv = dt[(long)m * D_INNER + d];
      float u   = xc[(long)m * D_INNER + d];
      float zv  = xz[(long)m * (2 * D_INNER) + D_INNER + d];
      float yv = 0.f;
#pragma unroll
      for (int n = 0; n < D_STATE; ++n) {
        hst[n] = __expf(dtv * a[n]) * hst[n] + dtv * u * bc[n];
        yv += hst[n] * bc[D_STATE + n];
      }
      yv += Dv * u;
      y[(long)m * D_INNER + d] = yv * siluf(zv);
    }
    __syncthreads();
  }
}

// ---------------------------------------------------------------------------
// Head: logits[b,c,l] = h[m,:]·head_w[c,:] + head_b[c]   (N_CLASSES=4)
// ---------------------------------------------------------------------------
__global__ __launch_bounds__(256)
void head_kernel(const float* __restrict__ hn, const float* __restrict__ w,
                 const float* __restrict__ bb, float* __restrict__ out)
{
  int m = blockIdx.x * 256 + threadIdx.x;
  if (m >= M_TOT) return;
  int b = m >> 11, l = m & (SEQLEN - 1);
  const float* row = hn + (long)m * D_MODEL;
  float acc[N_CLASSES] = {0.f, 0.f, 0.f, 0.f};
  for (int d = 0; d < D_MODEL; ++d) {
    float hv = row[d];
#pragma unroll
    for (int c = 0; c < N_CLASSES; ++c) acc[c] += hv * w[c * D_MODEL + d];
  }
#pragma unroll
  for (int c = 0; c < N_CLASSES; ++c)
    out[((long)b * N_CLASSES + c) * SEQLEN + l] = acc[c] + bb[c];
}

// ---------------------------------------------------------------------------
extern "C" void kernel_launch(void* const* d_in, const int* in_sizes, int n_in,
                              void* d_out, int out_size, void* d_ws, size_t ws_size,
                              hipStream_t stream)
{
  const float* x         = (const float*)d_in[0];
  const float* inp_w     = (const float*)d_in[1];
  const float* inp_b     = (const float*)d_in[2];
  const float* ln_w      = (const float*)d_in[3];
  const float* ln_b      = (const float*)d_in[4];
  const float* in_proj_w = (const float*)d_in[5];
  const float* conv_w    = (const float*)d_in[6];
  const float* conv_b    = (const float*)d_in[7];
  const float* x_proj_w  = (const float*)d_in[8];
  const float* dt_proj_w = (const float*)d_in[9];
  const float* dt_proj_b = (const float*)d_in[10];
  const float* A_log     = (const float*)d_in[11];
  const float* Dvec      = (const float*)d_in[12];
  const float* out_proj_w= (const float*)d_in[13];
  const float* fn_w      = (const float*)d_in[14];
  const float* fn_b      = (const float*)d_in[15];
  const float* head_w    = (const float*)d_in[16];
  const float* head_b    = (const float*)d_in[17];

  // workspace layout (floats)
  float* ws   = (float*)d_ws;
  float* hA    = ws;                                // 4096*512
  float* hB    = hA    + (long)M_TOT * D_MODEL;     // 4096*512
  float* hn    = hB    + (long)M_TOT * D_MODEL;     // 4096*512
  float* xz    = hn    + (long)M_TOT * D_MODEL;     // 4096*2048
  float* xc    = xz    + (long)M_TOT * 2 * D_INNER; // 4096*1024
  float* x_dbl = xc    + (long)M_TOT * D_INNER;     // 4096*64
  float* dt    = x_dbl + (long)M_TOT * 64;          // 4096*1024
  float* y     = dt    + (long)M_TOT * D_INNER;     // 4096*1024

  embed_kernel<<<(M_TOT * D_MODEL + 255) / 256, 256, 0, stream>>>(x, inp_w, inp_b, hA);

  float* hcur = hA;
  float* hnext = hB;
  for (int i = 0; i < N_LAYERS; ++i) {
    layernorm_kernel<<<M_TOT, 256, 0, stream>>>(
        hcur, ln_w + (long)i * D_MODEL, ln_b + (long)i * D_MODEL, hn);

    // in_proj: xz[4096,2048] = hn[4096,512] x W[2048,512]^T
    gemm_bf16_wmma<<<dim3(2 * D_INNER / BN, M_TOT / BM), 256, 0, stream>>>(
        hn, D_MODEL, in_proj_w + (long)i * 2 * D_INNER * D_MODEL, D_MODEL,
        xz, 2 * D_INNER, D_MODEL, 0, nullptr, nullptr);

    conv_silu_kernel<<<(M_TOT * D_INNER + 255) / 256, 256, 0, stream>>>(
        xz, conv_w + (long)i * D_INNER * D_CONV, conv_b + (long)i * D_INNER, xc);

    // x_proj: x_dbl[4096,64] = xc[4096,1024] x W[64,1024]^T
    gemm_bf16_wmma<<<dim3(1, M_TOT / BM), 256, 0, stream>>>(
        xc, D_INNER, x_proj_w + (long)i * 64 * D_INNER, D_INNER,
        x_dbl, 64, D_INNER, 0, nullptr, nullptr);

    // dt_proj: dt[4096,1024] = softplus(x_dbl[:, :32] x W[1024,32]^T + b)
    gemm_bf16_wmma<<<dim3(D_INNER / BN, M_TOT / BM), 256, 0, stream>>>(
        x_dbl, 64, dt_proj_w + (long)i * D_INNER * DT_RANK, DT_RANK,
        dt, D_INNER, DT_RANK, 1, dt_proj_b + (long)i * D_INNER, nullptr);

    scan_kernel<<<(BATCH * D_INNER) / 256, 256, 0, stream>>>(
        dt, x_dbl, xc, xz, A_log + (long)i * D_INNER * D_STATE,
        Dvec + (long)i * D_INNER, y);

    // out_proj + residual: hnext = y x W[512,1024]^T + hcur
    gemm_bf16_wmma<<<dim3(D_MODEL / BN, M_TOT / BM), 256, 0, stream>>>(
        y, D_INNER, out_proj_w + (long)i * D_MODEL * D_INNER, D_INNER,
        hnext, D_MODEL, D_INNER, 2, nullptr, hcur);

    float* tmp = hcur; hcur = hnext; hnext = tmp;
  }

  layernorm_kernel<<<M_TOT, 256, 0, stream>>>(hcur, fn_w, fn_b, hn);
  head_kernel<<<(M_TOT + 255) / 256, 256, 0, stream>>>(hn, head_w, head_b, (float*)d_out);
}
